// Retention_46961172414498
// MI455X (gfx1250) — compile-verified, matrix operand order checked by reading, not run on hardware
//
#include <hip/hip_runtime.h>
#include <cstdint>
#include <cstddef>

// ---------------------------------------------------------------------------
// Retention (decay-masked linear attention), MI455X / gfx1250, wave32 WMMA.
//   B=2, L=4096, H=1024, D=64.  gamma = 0.96875, xPos scale base 512.
// ---------------------------------------------------------------------------

#define RB 2
#define RL 4096
#define RH 1024
#define RD 64
#define WIN_TILES 24              // gamma^(23*64) ~ 5e-21 -> exact in fp32
#define LOG2_GAMMA (-0.0458039f)  // log2(0.96875)

typedef __attribute__((ext_vector_type(16))) __bf16 v16bf;
typedef __attribute__((ext_vector_type(8)))  float  v8f;

union FragAB { uint4 u[2]; v16bf v; };   // 16 bf16 = 32 B per lane

static __device__ inline v8f wmma_bf16(const v16bf& a, const v16bf& b, const v8f& c) {
  return __builtin_amdgcn_wmma_f32_16x16x32_bf16(false, a, false, b, (short)0, c, false, false);
}

// ---------------------------------------------------------------------------
// Tensor Data Mover (TDM) support
// ---------------------------------------------------------------------------
#if defined(__has_builtin)
#if __has_builtin(__builtin_amdgcn_tensor_load_to_lds) && __has_builtin(__builtin_amdgcn_s_wait_tensorcnt)
#define HAVE_TDM 1
#endif
#endif
#ifndef HAVE_TDM
#define HAVE_TDM 0
#endif

#if HAVE_TDM
typedef unsigned int tdm_v4u __attribute__((ext_vector_type(4)));
typedef int          tdm_v8i __attribute__((ext_vector_type(8)));
typedef int          tdm_v4i __attribute__((ext_vector_type(4)));

static __device__ inline unsigned ldsOff(const void* p) {
  // generic addr for LDS = aperture in [63:32] | byte offset in [31:0]
  return (unsigned)(unsigned long long)p;
}

// 2D tile load: 64x64 bf16 tile, hardware LDS padding 16B per 128B row
// -> LDS row stride 144 B (72 bf16) to avoid bank conflicts.
static __device__ inline void tdm_load_tile64(unsigned lds, const __bf16* g,
                                              unsigned td0, unsigned td1,
                                              unsigned stride0_elems) {
  unsigned long long ga = (unsigned long long)g;
  tdm_v4u g0 = { 1u,                                      // count=1, user mode
                 lds,                                     // lds_addr
                 (unsigned)ga,                            // global_addr[31:0]
                 ((unsigned)(ga >> 32) & 0x01FFFFFFu) | 0x80000000u }; // addr[56:32] | type=2
  // data_size=2B, pad_enable, pad_interval=32 dwords (128B), pad_amount=4 dwords (16B)
  unsigned w0 = (1u << 16) | (1u << 20) | (4u << 22) | (3u << 25);
  tdm_v8i g1 = { (int)w0,
                 (int)((td0 & 0xFFFFu) << 16),                    // tensor_dim0 lo
                 (int)((td0 >> 16) | ((td1 & 0xFFFFu) << 16)),    // tensor_dim0 hi | tensor_dim1 lo
                 (int)((td1 >> 16) | (64u << 16)),                // tensor_dim1 hi | tile_dim0=64
                 (int)64,                                         // tile_dim1=64, tile_dim2=0
                 (int)stride0_elems,                              // tensor_dim0_stride lo32
                 0, 0 };
  tdm_v4i z4 = { 0, 0, 0, 0 };
  tdm_v8i z8 = { 0, 0, 0, 0, 0, 0, 0, 0 };
  // 6-arg form (clang-23 / therock-10.0 headers): groups 0..3 + group for
  // gather/extended descriptors (zero = unused) + cache policy.
  __builtin_amdgcn_tensor_load_to_lds(g0, g1, z4, z4, z8, 0);
}
#else
static __device__ inline void copy_tile64(__bf16* dst /*stride 72*/, const __bf16* g,
                                          unsigned stride0_elems) {
  for (int i = threadIdx.x; i < 64 * 8; i += blockDim.x) {
    int r = i >> 3, c = (i & 7) * 8;
    *(uint4*)(dst + r * 72 + c) = *(const uint4*)(g + (size_t)r * stride0_elems + c);
  }
}
#endif

// ---------------------------------------------------------------------------
// Kernel 0: convert W_Q/W_K/W_V (fp32 1024x64) to bf16 in WMMA-B-fragment order.
// Fragment index t = mat*4 + coltile. Layout: [kc(32)][t(12)][lane(32)][16 bf16]
// lane<16: n=lane, k = kc*32 + j (j<16) ; lane>=16: same n, k = kc*32+16+j
// ---------------------------------------------------------------------------
__global__ __launch_bounds__(256) void pack_w(const float* __restrict__ WQ,
                                              const float* __restrict__ WK,
                                              const float* __restrict__ WV,
                                              unsigned short* __restrict__ wfrag_) {
  __bf16* wfrag = (__bf16*)wfrag_;
  int tid = blockIdx.x * blockDim.x + threadIdx.x;
  if (tid >= 32 * 12 * 32) return;
  int lane = tid & 31;
  int t    = (tid >> 5) % 12;
  int kc   = (tid >> 5) / 12;
  int mat  = t >> 2, tc = t & 3;
  const float* Wsrc = (mat == 0) ? WQ : (mat == 1) ? WK : WV;
  int k0 = kc * 32 + (lane >> 4) * 16;
  int n  = tc * 16 + (lane & 15);
  __bf16* dst = wfrag + (((size_t)kc * 12 + t) * 32 + lane) * 16;
#pragma unroll
  for (int j = 0; j < 16; ++j)
    dst[j] = (__bf16)Wsrc[(size_t)(k0 + j) * RD + n];
}

// ---------------------------------------------------------------------------
// Kernel 1: fused projections + xPos.  One wave = 16 rows of X; one A-fragment
// per K-chunk feeds 12 WMMAs (Q,K,V x 4 column tiles).  Epilogue applies xPos
// (rotate pair via shfl_xor) and writes Q,K row-major bf16 and V transposed.
// ---------------------------------------------------------------------------
__global__ __launch_bounds__(64) void proj_xpos(const float* __restrict__ X,
                                                const unsigned short* __restrict__ wfrag_,
                                                unsigned short* __restrict__ Qb_,
                                                unsigned short* __restrict__ Kb_,
                                                unsigned short* __restrict__ Vt_) {
  const __bf16* wfrag = (const __bf16*)wfrag_;
  __bf16* Qb = (__bf16*)Qb_;
  __bf16* Kb = (__bf16*)Kb_;
  __bf16* Vt = (__bf16*)Vt_;

  const int wv = threadIdx.x >> 5, lane = threadIdx.x & 31;
  const int ln = lane & 15, lh = lane >> 4;
  const int m0 = blockIdx.x * 32 + wv * 16;            // 8192/32 = 256 blocks

  v8f acc[12];
#pragma unroll
  for (int t = 0; t < 12; ++t) acc[t] = (v8f){0.f,0.f,0.f,0.f,0.f,0.f,0.f,0.f};

  for (int kc = 0; kc < 32; ++kc) {
    const float* xr = X + (size_t)(m0 + ln) * RH + kc * 32 + lh * 16;
    if (kc + 1 < 32) __builtin_prefetch(xr + 32, 0, 3);   // global_prefetch_b8
    __attribute__((aligned(16))) float xf[16];
    *(float4*)&xf[0]  = ((const float4*)xr)[0];
    *(float4*)&xf[4]  = ((const float4*)xr)[1];
    *(float4*)&xf[8]  = ((const float4*)xr)[2];
    *(float4*)&xf[12] = ((const float4*)xr)[3];
    v16bf a;
#pragma unroll
    for (int j = 0; j < 16; ++j) a[j] = (__bf16)xf[j];

#pragma unroll
    for (int t = 0; t < 12; ++t) {
      FragAB bw;
      const uint4* wp = (const uint4*)(wfrag + (((size_t)kc * 12 + t) * 32 + lane) * 16);
      bw.u[0] = wp[0];
      bw.u[1] = wp[1];
      acc[t] = wmma_bf16(a, bw.v, acc[t]);
    }
  }

  // Epilogue: accumulator element e -> row M = e + 8*(lane>=16), col = tc*16 + ln
#pragma unroll
  for (int t = 0; t < 12; ++t) {
    const int mat = t >> 2, tc = t & 3;
#pragma unroll
    for (int e = 0; e < 8; ++e) {
      float q = acc[t][e];
      float p = __shfl_xor(q, 1, 32);                 // rotate-pair partner (adjacent col)
      int m = m0 + e + 8 * lh;                        // flat row in [0, B*L)
      int b = m >> 12;                                // / 4096
      int l = m & (RL - 1);
      int j = tc * 16 + ln;
      if (mat == 2) {
        Vt[((size_t)b * RD + j) * RL + l] = (__bf16)q;  // V stored transposed
      } else {
        float rot = (j & 1) ? p : -p;                 // rotate_every_two
        int   i   = j >> 1;
        float invf = exp2f(-0.41524101f * (float)i);  // 10000^(-i/32)
        float ph   = (float)l * invf;
        float sn = __sinf(ph), cs = __cosf(ph);
        float sv = (2.0f * (float)i + 25.6f) * (1.0f / 89.6f);
        float ex = log2f(sv) * ((float)l * (1.0f / 512.0f));
        float scl = exp2f((mat == 0) ? ex : -ex);     // K uses downscale
        float val = (q * cs + rot * sn) * scl;
        __bf16* dst = (mat == 0) ? Qb : Kb;
        dst[(size_t)m * RD + j] = (__bf16)val;
      }
    }
  }
}

// ---------------------------------------------------------------------------
// Kernel 2: windowed retention.  One WG = 64 output rows (4 waves x 16 rows).
// Double-buffered TDM loads of K/V 64x64 bf16 tiles into padded LDS.
//   S = Q K^T (WMMA), elementwise *= gamma^(n-m) (0 above diagonal),
//   C-layout -> A-layout via LDS, O += S V (WMMA).
// ---------------------------------------------------------------------------
__global__ __launch_bounds__(128) void retention_attn(const unsigned short* __restrict__ Qb_,
                                                      const unsigned short* __restrict__ Kb_,
                                                      const unsigned short* __restrict__ Vt_,
                                                      float* __restrict__ out) {
  const __bf16* Qb = (const __bf16*)Qb_;
  const __bf16* Kb = (const __bf16*)Kb_;
  const __bf16* Vt = (const __bf16*)Vt_;

  __shared__ __attribute__((aligned(16))) __bf16 Ksh[2][64][72];
  __shared__ __attribute__((aligned(16))) __bf16 Vsh[2][64][72];
  __shared__ __attribute__((aligned(16))) __bf16 Ssh[4][16][72];

  const int wv = threadIdx.x >> 5, lane = threadIdx.x & 31;
  const int ln = lane & 15, lh = lane >> 4;
  const int rt = blockIdx.x, b = blockIdx.y;
  const int rowbase = rt * 64 + wv * 16;

  // Q A-fragments (16x64 bf16 -> two 16x32 frags), rows L2-resident
  FragAB aQ[2];
  const __bf16* qrow = Qb + ((size_t)b * RL + rowbase + ln) * RD;
#pragma unroll
  for (int h = 0; h < 2; ++h) {
    const uint4* p = (const uint4*)(qrow + h * 32 + lh * 16);
    aQ[h].u[0] = p[0];
    aQ[h].u[1] = p[1];
  }

  v8f accO[4];
#pragma unroll
  for (int oc = 0; oc < 4; ++oc) accO[oc] = (v8f){0.f,0.f,0.f,0.f,0.f,0.f,0.f,0.f};

  const int ct0 = (rt > WIN_TILES - 1) ? rt - (WIN_TILES - 1) : 0;

#if HAVE_TDM
  if (wv == 0) {   // one wave drives the TDM; TENSORcnt is per-wave
    tdm_load_tile64(ldsOff(&Ksh[0][0][0]),
                    Kb + ((size_t)b * RL + (size_t)ct0 * 64) * RD, RD, RL, RD);
    tdm_load_tile64(ldsOff(&Vsh[0][0][0]),
                    Vt + (size_t)b * RD * RL + (size_t)ct0 * 64, RL, RD, RL);
  }
#endif

  for (int ct = ct0; ct <= rt; ++ct) {
    const int buf = (ct - ct0) & 1;
    __syncthreads();   // previous consumers of buf^1 are done
#if HAVE_TDM
    if (wv == 0) {
      if (ct < rt) {   // prefetch next tile, then wait for current
        tdm_load_tile64(ldsOff(&Ksh[buf ^ 1][0][0]),
                        Kb + ((size_t)b * RL + (size_t)(ct + 1) * 64) * RD, RD, RL, RD);
        tdm_load_tile64(ldsOff(&Vsh[buf ^ 1][0][0]),
                        Vt + (size_t)b * RD * RL + (size_t)(ct + 1) * 64, RL, RD, RL);
        __builtin_amdgcn_s_wait_tensorcnt(2);
      } else {
        __builtin_amdgcn_s_wait_tensorcnt(0);
      }
    }
#else
    copy_tile64(&Ksh[buf][0][0], Kb + ((size_t)b * RL + (size_t)ct * 64) * RD, RD);
    copy_tile64(&Vsh[buf][0][0], Vt + (size_t)b * RD * RL + (size_t)ct * 64, RL);
#endif
    __syncthreads();   // publish tile ct

    // ---- S = Q K^T over four 16x16 score tiles, decay, park in LDS (bf16) ----
#pragma unroll
    for (int st = 0; st < 4; ++st) {
      FragAB bk0, bk1;   // B = K^T: lane n holds K-row (ct*64 + st*16 + n), contiguous d
      const __bf16* kr = &Ksh[buf][st * 16 + ln][0];
      {
        const uint4* p0 = (const uint4*)(kr + lh * 16);
        const uint4* p1 = (const uint4*)(kr + 32 + lh * 16);
        bk0.u[0] = p0[0]; bk0.u[1] = p0[1];
        bk1.u[0] = p1[0]; bk1.u[1] = p1[1];
      }
      v8f s = (v8f){0.f,0.f,0.f,0.f,0.f,0.f,0.f,0.f};
      s = wmma_bf16(aQ[0].v, bk0.v, s);
      s = wmma_bf16(aQ[1].v, bk1.v, s);
#pragma unroll
      for (int e = 0; e < 8; ++e) {
        int n = rowbase + e + 8 * lh;
        int m = ct * 64 + st * 16 + ln;
        int diff = n - m;
        float w = (diff >= 0) ? exp2f((float)diff * LOG2_GAMMA) : 0.0f;  // v_exp_f32
        Ssh[wv][e + 8 * lh][st * 16 + ln] = (__bf16)(s[e] * w);
      }
    }

    // ---- reload S as A-fragments (per-wave, DS in-order) and do O += S V ----
    FragAB aS[2];
    const __bf16* srow = &Ssh[wv][ln][0];
#pragma unroll
    for (int h = 0; h < 2; ++h) {
      const uint4* p = (const uint4*)(srow + h * 32 + lh * 16);
      aS[h].u[0] = p[0];
      aS[h].u[1] = p[1];
    }
#pragma unroll
    for (int oc = 0; oc < 4; ++oc) {
      FragAB bv0, bv1;   // B = V: lane n holds V^T row (col j = oc*16+n), contiguous m
      const __bf16* vr = &Vsh[buf][oc * 16 + ln][0];
      {
        const uint4* p0 = (const uint4*)(vr + lh * 16);
        const uint4* p1 = (const uint4*)(vr + 32 + lh * 16);
        bv0.u[0] = p0[0]; bv0.u[1] = p0[1];
        bv1.u[0] = p1[0]; bv1.u[1] = p1[1];
      }
      accO[oc] = wmma_bf16(aS[0].v, bv0.v, accO[oc]);
      accO[oc] = wmma_bf16(aS[1].v, bv1.v, accO[oc]);
    }
  }

  // ---- store O (fp32) ----
#pragma unroll
  for (int oc = 0; oc < 4; ++oc)
#pragma unroll
    for (int e = 0; e < 8; ++e) {
      int n = rowbase + e + 8 * lh;
      out[((size_t)b * RL + n) * RD + oc * 16 + ln] = accO[oc][e];
    }
}

// ---------------------------------------------------------------------------
extern "C" void kernel_launch(void* const* d_in, const int* in_sizes, int n_in,
                              void* d_out, int out_size, void* d_ws, size_t ws_size,
                              hipStream_t stream) {
  const float* X  = (const float*)d_in[0];
  const float* WQ = (const float*)d_in[1];
  const float* WK = (const float*)d_in[2];
  const float* WV = (const float*)d_in[3];
  float* out = (float*)d_out;

  // workspace layout (bf16 as ushort): Wfrag 384 KB | Qb 1 MB | Kb 1 MB | Vt 1 MB
  unsigned short* wfrag = (unsigned short*)d_ws;
  unsigned short* Qb = (unsigned short*)((char*)d_ws + 393216);
  unsigned short* Kb = Qb + (size_t)RB * RL * RD;
  unsigned short* Vt = Kb + (size_t)RB * RL * RD;

  pack_w<<<48, 256, 0, stream>>>(WQ, WK, WV, wfrag);
  proj_xpos<<<256, 64, 0, stream>>>(X, wfrag, Qb, Kb, Vt);
  retention_attn<<<dim3(RL / 64, RB), 128, 0, stream>>>(Qb, Kb, Vt, out);
}
